// DGL_mean_Classifier_75333726371912
// MI455X (gfx1250) — matrix operand order repaired and checked by reading, not compiled, not run on hardware
//
#include <hip/hip_runtime.h>
#include <hip/hip_bf16.h>
#include <stdint.h>

#define NN 100000
#define EE 1600000
#define GG 128

typedef __attribute__((ext_vector_type(16))) __bf16         v16bf;
typedef __attribute__((ext_vector_type(8)))  float          v8f;
typedef __attribute__((ext_vector_type(8)))  unsigned short ushort8;
typedef __attribute__((ext_vector_type(16))) unsigned short ushort16;
typedef __attribute__((ext_vector_type(4)))  unsigned int   tdm_g0_t;
typedef __attribute__((ext_vector_type(8)))  int            tdm_g1_t;
typedef __attribute__((ext_vector_type(4)))  int            tdm_g2_t;

#if defined(__has_builtin)
#  if __has_builtin(__builtin_amdgcn_tensor_load_to_lds)
#    define HAVE_TDM 1
#  endif
#endif
#ifndef HAVE_TDM
#  define HAVE_TDM 0
#endif

#if HAVE_TDM
__device__ __forceinline__ void tdm_load(tdm_g0_t g0, tdm_g1_t g1) {
  tdm_g2_t z4 = {0, 0, 0, 0};
#if __clang_major__ >= 23
  tdm_g1_t z8 = {0, 0, 0, 0, 0, 0, 0, 0};
  __builtin_amdgcn_tensor_load_to_lds(g0, g1, z4, z4, z8, 0);
#else
  __builtin_amdgcn_tensor_load_to_lds(g0, g1, z4, z4, 0);
#endif
}
#endif

__device__ __forceinline__ unsigned short f32_bf16(float x) {
  unsigned int u = __float_as_uint(x);
  u += 0x7FFFu + ((u >> 16) & 1u);   // round-to-nearest-even
  return (unsigned short)(u >> 16);
}

// ---------------- degree / norm ----------------
__global__ void deg_kernel(const int* __restrict__ dst, float* __restrict__ deg, int E) {
  int e = blockIdx.x * blockDim.x + threadIdx.x;
  if (e < E) atomicAdd(&deg[dst[e]], 1.0f);
}

__global__ void norm_kernel(float* __restrict__ norm, int n) {
  int i = blockIdx.x * blockDim.x + threadIdx.x;
  if (i < n) {
    float d = norm[i];
    d = d < 1.0f ? 1.0f : d;
    norm[i] = rsqrtf(d);
  }
}

// ---------------- bf16 conversion ----------------
__global__ void cvt_bf16(const float* __restrict__ x, unsigned short* __restrict__ y, int n) {
  int i = blockIdx.x * blockDim.x + threadIdx.x;
  if (i < n) y[i] = f32_bf16(x[i]);
}

__global__ void pack_bf16_seg(const float* __restrict__ X, unsigned short* __restrict__ xcat,
                              int F, int Kt, int seg) {
  int row = blockIdx.x, f = threadIdx.x;
  xcat[(size_t)row * Kt + seg * F + f] = f32_bf16(X[(size_t)row * F + f]);
}

// ---------------- SpMM: T[dst] += (h*norm)[src] ----------------
__global__ __launch_bounds__(256)
void spmm_scatter(const float* __restrict__ h, const float* __restrict__ norm,
                  const int* __restrict__ src, const int* __restrict__ dst,
                  float* __restrict__ T, int F, int E) {
  int lane = threadIdx.x & 31;
  int eloc = threadIdx.x >> 5;
  int e = blockIdx.x * 8 + eloc;
  if (e >= E) return;
  int s = src[e], d = dst[e];
  float ns = norm[s];
  const float4* hs = (const float4*)(h + (size_t)s * F);
  float* td = T + (size_t)d * F;
  int nch = F >> 7;
  for (int c = 0; c < nch; ++c) {
    float4 v = hs[c * 32 + lane];
    int fo = (c * 32 + lane) * 4;
    atomicAdd(&td[fo + 0], v.x * ns);
    atomicAdd(&td[fo + 1], v.y * ns);
    atomicAdd(&td[fo + 2], v.z * ns);
    atomicAdd(&td[fo + 3], v.w * ns);
  }
}

__global__ void combine1(float* __restrict__ T, const float* __restrict__ norm,
                         unsigned short* __restrict__ xcat, int F, int Kt, int seg) {
  int row = blockIdx.x, f = threadIdx.x;
  size_t i = (size_t)row * F + f;
  float v = -norm[row] * T[i];
  T[i] = v;
  xcat[(size_t)row * Kt + seg * F + f] = f32_bf16(v);
}

__global__ void combine2(const float* __restrict__ U, const float* __restrict__ h,
                         const float* __restrict__ norm, unsigned short* __restrict__ xcat,
                         int F, int Kt, int seg) {
  int row = blockIdx.x, f = threadIdx.x;
  size_t i = (size_t)row * F + f;
  float v = -2.0f * norm[row] * U[i] - h[i];
  xcat[(size_t)row * Kt + seg * F + f] = f32_bf16(v);
}

// ---------------- WMMA GEMM: C = act(A(bf16) @ B(bf16) + bias) ----------------
// Block = 256 threads (8 waves). Tile BM=256 x BN=64, K-step 32.
// Wave w computes rows [w*32, w*32+32) x 64 cols: 2 A-frags x 4 B-frags = 8 WMMA / k-tile.
// A tile staged by Tensor Data Mover (async, LDS row padding via pad_interval/amount),
// B tile staged by threads (transposed, K-contiguous). Double-buffered LDS.
#define BM 256
#define BN 64
#define BK 32
#define LDA_S 40   // 32 data + 8 pad ushorts (TDM: pad 4 DW every 16 DW)
#define LDB_S 40

__global__ void __launch_bounds__(256)
wmma_gemm_bias_act(const unsigned short* __restrict__ A,
                   const unsigned short* __restrict__ B,
                   const float* __restrict__ bias,
                   float* __restrict__ C,
                   int M, int Kt, int Nout, int doRelu) {
  __shared__ __align__(16) unsigned short As[2][BM * LDA_S];  // 2 x 20 KB
  __shared__ __align__(16) unsigned short Bs[2][BN * LDB_S];  // 2 x 5 KB

  const int tid  = threadIdx.x;
  const int lane = tid & 31;
  const int wave = tid >> 5;               // 0..7
  const int khalf = lane >> 4;             // K-half selector
  const int l15   = lane & 15;
  const int rowBlock = blockIdx.x * BM;
  const int colBase  = blockIdx.y * BN;

  v8f acc[2][4];
#pragma unroll
  for (int n = 0; n < 4; ++n) {
    float bv = bias[colBase + n * 16 + l15];
#pragma unroll
    for (int s = 0; s < 2; ++s)
#pragma unroll
      for (int r = 0; r < 8; ++r) acc[s][n][r] = bv;
  }

  const int bkr = tid >> 3;                // B tile k-row 0..31
  const int bc8 = tid & 7;                 // B col octet 0..7
  const int nK = Kt / BK;

#if HAVE_TDM
  // --- TDM descriptor issue: 2D tile BK x BM (bf16), row stride Kt, OOB rows -> 0 ---
  auto issueA = [&](int kt, int buf) {
    unsigned long long gaddr = (unsigned long long)(const void*)A +
        (((unsigned long long)rowBlock * (unsigned)Kt + (unsigned)(kt * BK)) << 1);
    unsigned int lds = (unsigned int)(unsigned long long)&As[buf][0];
    tdm_g0_t g0;
    g0[0] = 1u;                                     // count=1, user descriptor
    g0[1] = lds;                                    // lds_addr
    g0[2] = (unsigned int)gaddr;                    // global_addr[31:0]
    g0[3] = (unsigned int)(gaddr >> 32) | (2u << 30); // global_addr[56:32] | type=2
    unsigned int td0 = (unsigned int)Kt;            // tensor_dim0 (elements)
    unsigned int td1 = (unsigned int)M;             // tensor_dim1 (rows; OOB -> 0)
    unsigned long long st0 = (unsigned long long)(unsigned)Kt; // dim0 stride
    tdm_g1_t g1;
    g1[0] = (int)((1u << 16)     // data_size = 2 bytes
                | (1u << 20)     // pad_enable
                | (3u << 22)     // pad_interval: every 16 DWORDs
                | (3u << 25));   // pad_amount: 4 DWORDs
    g1[1] = (int)(td0 << 16);                            // tensor_dim0[15:0]
    g1[2] = (int)((td0 >> 16) | (td1 << 16));            // dim0[31:16] | dim1[15:0]
    g1[3] = (int)((td1 >> 16) | ((unsigned)BK << 16));   // dim1[31:16] | tile_dim0=32
    g1[4] = (int)((unsigned)BM);                         // tile_dim1=256, tile_dim2=0
    g1[5] = (int)(unsigned int)(st0 & 0xFFFFFFFFu);      // dim0_stride[31:0]
    g1[6] = (int)((unsigned int)(st0 >> 32) & 0xFFFFu);  // dim0_stride[47:32]
    g1[7] = 0;
    tdm_load(g0, g1);
  };
#else
  // fallback: thread-staged A tile (row per thread, 64B)
  auto issueA = [&](int kt, int buf) {
    int grow = rowBlock + tid; if (grow >= M) grow = M - 1;
    const ushort8* s = (const ushort8*)(A + (size_t)grow * Kt + kt * BK);
#pragma unroll
    for (int q = 0; q < 4; ++q)
      *(ushort8*)&As[buf][tid * LDA_S + q * 8] = s[q];
  };
#endif

  ushort8 breg;
  auto loadB = [&](int kt) {
    breg = *(const ushort8*)(B + (size_t)(kt * BK + bkr) * Nout + colBase + bc8 * 8);
  };
  auto storeB = [&](int buf) {
#pragma unroll
    for (int j = 0; j < 8; ++j)
      Bs[buf][(bc8 * 8 + j) * LDB_S + bkr] = breg[j];
  };

  // prologue: stage tile 0
#if HAVE_TDM
  if (wave == 0) issueA(0, 0);
#else
  issueA(0, 0);
#endif
  loadB(0);
  storeB(0);

  for (int kt = 0; kt < nK; ++kt) {
    const int cur = kt & 1, nxt = cur ^ 1;
    const bool more = (kt + 1 < nK);
    if (more) {
#if HAVE_TDM
      if (wave == 0) issueA(kt + 1, nxt);   // async into other buffer
#else
      issueA(kt + 1, nxt);
#endif
      loadB(kt + 1);                        // global loads in flight during compute
    }
#if HAVE_TDM
    if (wave == 0) {                        // TENSORcnt is per-issuing-wave
      if (more) __builtin_amdgcn_s_wait_tensorcnt(1);
      else      __builtin_amdgcn_s_wait_tensorcnt(0);
    }
#endif
    __syncthreads();                        // publish As[cur] + Bs[cur]

#pragma unroll
    for (int s = 0; s < 2; ++s) {
      const int arow = wave * 32 + s * 16 + l15;
      ushort8 alo = *(const ushort8*)&As[cur][arow * LDA_S + khalf * 8];
      ushort8 ahi = *(const ushort8*)&As[cur][arow * LDA_S + 16 + khalf * 8];
      ushort16 av;
#pragma unroll
      for (int j = 0; j < 8; ++j) { av[j] = alo[j]; av[j + 8] = ahi[j]; }
      v16bf afrag = __builtin_bit_cast(v16bf, av);
#pragma unroll
      for (int n = 0; n < 4; ++n) {
        const int bcol = n * 16 + l15;
        ushort8 blo = *(const ushort8*)&Bs[cur][bcol * LDB_S + khalf * 8];
        ushort8 bhi = *(const ushort8*)&Bs[cur][bcol * LDB_S + 16 + khalf * 8];
        ushort16 bv;
#pragma unroll
        for (int j = 0; j < 8; ++j) { bv[j] = blo[j]; bv[j + 8] = bhi[j]; }
        v16bf bfrag = __builtin_bit_cast(v16bf, bv);
        acc[s][n] = __builtin_amdgcn_wmma_f32_16x16x32_bf16(
            false, afrag, false, bfrag, (short)0, acc[s][n], false, false);
      }
    }
    if (more) storeB(nxt);                  // deferred B store into other buffer
    __syncthreads();                        // readers done before next tile's writes
  }

  // epilogue
#pragma unroll
  for (int s = 0; s < 2; ++s) {
    const int mbase = rowBlock + wave * 32 + s * 16 + khalf * 8;
#pragma unroll
    for (int n = 0; n < 4; ++n) {
#pragma unroll
      for (int r = 0; r < 8; ++r) {
        int gr = mbase + r;
        if (gr < M) {
          float v = acc[s][n][r];
          if (doRelu) v = v > 0.0f ? v : 0.0f;
          C[(size_t)gr * Nout + colBase + n * 16 + l15] = v;
        }
      }
    }
  }
}

// ---------------- graph pooling ----------------
__global__ void pool_kernel(const float* __restrict__ h, const int* __restrict__ gid,
                            float* __restrict__ hg, int F) {
  int row = blockIdx.x, f = threadIdx.x;     // blockDim = 512
  int g = gid[row];
  atomicAdd(&hg[(size_t)g * F + f], h[(size_t)row * F + f]);
}

// ---------------- tiny final GEMM (128x512x10, no relu) ----------------
__global__ void gemv_small(const float* __restrict__ X, const float* __restrict__ W,
                           const float* __restrict__ b, float* __restrict__ out,
                           int M, int Kt, int Nout) {
  int i = blockIdx.x * blockDim.x + threadIdx.x;
  if (i >= M * Nout) return;
  int m = i / Nout, o = i % Nout;
  float s = b[o];
  for (int k = 0; k < Kt; ++k) s += X[(size_t)m * Kt + k] * W[(size_t)k * Nout + o];
  out[i] = s;
}

extern "C" void kernel_launch(void* const* d_in, const int* in_sizes, int n_in,
                              void* d_out, int out_size, void* d_ws, size_t ws_size,
                              hipStream_t stream) {
  (void)in_sizes; (void)n_in; (void)out_size; (void)ws_size;
  const float* signal = (const float*)d_in[0];
  const float* W[4]   = {(const float*)d_in[1], (const float*)d_in[3],
                         (const float*)d_in[5], (const float*)d_in[7]};
  const float* Bv[4]  = {(const float*)d_in[2], (const float*)d_in[4],
                         (const float*)d_in[6], (const float*)d_in[8]};
  const float* Wm1 = (const float*)d_in[9];
  const float* bm1 = (const float*)d_in[10];
  const float* Wm2 = (const float*)d_in[11];
  const float* bm2 = (const float*)d_in[12];
  const int* src = (const int*)d_in[13];
  const int* dst = (const int*)d_in[14];
  const int* gid = (const int*)d_in[15];
  float* out = (float*)d_out;

  char* ws = (char*)d_ws;
  size_t off = 0;
  auto carve = [&](size_t bytes) -> void* {
    off = (off + 255) & ~(size_t)255;
    void* p = ws + off; off += bytes; return p;
  };
  float* norm = (float*)carve((size_t)NN * 4);
  float* hA   = (float*)carve((size_t)NN * 512 * 4);
  float* hB   = (float*)carve((size_t)NN * 512 * 4);
  float* T    = (float*)carve((size_t)NN * 256 * 4);
  float* U    = (float*)carve((size_t)NN * 256 * 4);
  unsigned short* Xcat = (unsigned short*)carve((size_t)NN * 768 * 2);
  int Fin[4]  = {128, 128, 128, 256};
  int Fout[4] = {128, 128, 256, 512};
  unsigned short* Wb[4];
  for (int l = 0; l < 4; ++l)
    Wb[l] = (unsigned short*)carve((size_t)3 * Fin[l] * Fout[l] * 2);
  unsigned short* WbM1 = (unsigned short*)carve((size_t)512 * 512 * 2);
  float* hg  = (float*)carve((size_t)GG * 512 * 4);
  unsigned short* hgb = (unsigned short*)carve((size_t)GG * 512 * 2);
  float* mlp1 = (float*)carve((size_t)GG * 512 * 4);

  // degree -> norm = clip(deg,1)^-0.5
  hipMemsetAsync(norm, 0, (size_t)NN * 4, stream);
  deg_kernel<<<(EE + 255) / 256, 256, 0, stream>>>(dst, norm, EE);
  norm_kernel<<<(NN + 255) / 256, 256, 0, stream>>>(norm, NN);

  // weights -> bf16
  for (int l = 0; l < 4; ++l) {
    int n = 3 * Fin[l] * Fout[l];
    cvt_bf16<<<(n + 255) / 256, 256, 0, stream>>>(W[l], Wb[l], n);
  }
  cvt_bf16<<<(512 * 512 + 255) / 256, 256, 0, stream>>>(Wm1, WbM1, 512 * 512);

  const float* hcur = signal;
  float* pong[2] = {hA, hB};
  for (int l = 0; l < 4; ++l) {
    int F = Fin[l], Kt = 3 * F, Fo = Fout[l];
    pack_bf16_seg<<<NN, F, 0, stream>>>(hcur, Xcat, F, Kt, 0);
    // X1 = -Ahat X0
    hipMemsetAsync(T, 0, (size_t)NN * F * 4, stream);
    spmm_scatter<<<(EE + 7) / 8, 256, 0, stream>>>(hcur, norm, src, dst, T, F, EE);
    combine1<<<NN, F, 0, stream>>>(T, norm, Xcat, F, Kt, 1);
    // X2 = -2 Ahat X1 - X0
    hipMemsetAsync(U, 0, (size_t)NN * F * 4, stream);
    spmm_scatter<<<(EE + 7) / 8, 256, 0, stream>>>(T, norm, src, dst, U, F, EE);
    combine2<<<NN, F, 0, stream>>>(U, hcur, norm, Xcat, F, Kt, 2);
    // h = relu([X0|X1|X2] @ W + b)
    float* hnext = pong[l & 1];
    dim3 g((NN + BM - 1) / BM, Fo / BN);
    wmma_gemm_bias_act<<<g, 256, 0, stream>>>(Xcat, Wb[l], Bv[l], hnext, NN, Kt, Fo, 1);
    hcur = hnext;
  }

  // graph pooling + MLP
  hipMemsetAsync(hg, 0, (size_t)GG * 512 * 4, stream);
  pool_kernel<<<NN, 512, 0, stream>>>(hcur, gid, hg, 512);
  cvt_bf16<<<(GG * 512 + 255) / 256, 256, 0, stream>>>(hg, hgb, GG * 512);
  {
    dim3 g(1, 512 / BN);
    wmma_gemm_bias_act<<<g, 256, 0, stream>>>(hgb, WbM1, bm1, mlp1, GG, 512, 512, 1);
  }
  gemv_small<<<(GG * 10 + 255) / 256, 256, 0, stream>>>(mlp1, Wm2, bm2, out, GG, 512, 10);
}
